// BiMinGRU_80470507257986
// MI455X (gfx1250) — matrix operand order repaired and checked by reading, not compiled, not run on hardware
//
#include <hip/hip_runtime.h>
#include <math.h>

// ---------------------------------------------------------------------------
// Bidirectional minGRU for MI455X (gfx1250), bf16 WMMA GEMMs + f32 log-space
// chunked scan.  B=4, S=4096, D=H=1024.
// ---------------------------------------------------------------------------

typedef __bf16 bf16_t;
typedef bf16_t v16bf __attribute__((ext_vector_type(16)));
typedef float  v8f   __attribute__((ext_vector_type(8)));
typedef float  v4f   __attribute__((ext_vector_type(4)));

#define Bsz 4
#define Ssz 4096
#define Dsz 1024
#define Hsz 1024
#define Msz (Bsz * Ssz)     // 16384
#define K2  (2 * Hsz)       // 2048
#define NCH 16              // scan chunks along S
#define CHS (Ssz / NCH)     // 256

// ---------------------------------------------------------------------------
// math helpers (f32, numerically stable)
// ---------------------------------------------------------------------------
__device__ __forceinline__ float softplusf(float x) {
    return fmaxf(x, 0.0f) + log1pf(__expf(-fabsf(x)));
}
__device__ __forceinline__ float log_g(float x) {
    return (x >= 0.0f) ? __logf(x + 0.5f) : -softplusf(-x);
}
__device__ __forceinline__ float logaddexpf_(float a, float b) {
    float m = fmaxf(a, b);
    if (m == -INFINITY) return -INFINITY;
    return m + log1pf(__expf(fminf(a, b) - m));
}

// ---------------------------------------------------------------------------
// WMMA fragment helpers
// ---------------------------------------------------------------------------
__device__ __forceinline__ v8f wmma_bf16(v16bf a, v16bf b, v8f c) {
    // D = A(16x32 bf16) * B(32x16 bf16) + C(16x16 f32)
    return __builtin_amdgcn_wmma_f32_16x16x32_bf16(
        false, a, false, b, (short)0, c, false, false);
}

// A-matrix 16x32 bf16 per ISA table: lane<16 holds M=lane, K in {0..7,16..23};
// lane>=16 holds M=lane-16, K in {8..15,24..31}.  From a row-major bf16 row
// pointer this is two contiguous 16-byte loads.
__device__ __forceinline__ v16bf load_fragA_row(const bf16_t* rowptr, int kt, int kh) {
    const bf16_t* p = rowptr + kt + kh * 8;
    union { v16bf v; v4f f[2]; } u;
    u.f[0] = *(const v4f*)(p);
    u.f[1] = *(const v4f*)(p + 16);
    return u.v;
}

// B-matrix 32x16 bf16: lanes 0..15 hold K in {0..7,16..23} of column N=lane, lanes 16..31
// hold K in {8..15,24..31} (matches A).  Weights are pre-transposed to
// [N][K] row-major so this is two contiguous 16-byte loads per lane.
__device__ __forceinline__ v16bf load_fragB(const bf16_t* Wt, int ldk, int col0,
                                            int kt, int lane) {
    const bf16_t* p = Wt + (size_t)(col0 + (lane & 15)) * ldk + kt + (lane >> 4) * 8;
    union { v16bf v; v4f f[2]; } u;
    u.f[0] = *(const v4f*)(p);
    u.f[1] = *(const v4f*)(p + 16);
    return u.v;
}

// ---------------------------------------------------------------------------
// prep kernels: f32 -> bf16 convert, and transpose+convert for weights
// ---------------------------------------------------------------------------
__global__ __launch_bounds__(256)
void convert_bf16_kernel(const float* __restrict__ src, bf16_t* __restrict__ dst,
                         size_t n4) {
    size_t i = (size_t)blockIdx.x * blockDim.x + threadIdx.x;
    if (i >= n4) return;
    v4f f = *(const v4f*)(src + i * 4);
    bf16_t* d = dst + i * 4;
    d[0] = (bf16_t)f.x; d[1] = (bf16_t)f.y; d[2] = (bf16_t)f.z; d[3] = (bf16_t)f.w;
}

// Wt[n][k] = bf16(W[k][n]);  W is [K][N] row-major.
__global__ __launch_bounds__(256)
void transconv_kernel(const float* __restrict__ W, bf16_t* __restrict__ Wt,
                      int K, int N) {
    __shared__ float tile[32][33];
    int k0 = blockIdx.y * 32, n0 = blockIdx.x * 32;
    int tx = threadIdx.x, ty = threadIdx.y;   // block (32,8)
    for (int i = ty; i < 32; i += 8)
        tile[i][tx] = W[(size_t)(k0 + i) * N + (n0 + tx)];
    __syncthreads();
    for (int i = ty; i < 32; i += 8)
        Wt[(size_t)(n0 + i) * K + (k0 + tx)] = (bf16_t)tile[tx][i];
}

// ---------------------------------------------------------------------------
// Gate GEMM:  k = x@Wz + bz, hpre = x@Wh + bh, fused epilogue producing
//   lc  = -softplus(k)               (log(1-z))
//   lzh = -softplus(-k) + log_g(hpre)  (log z + log h~)
// Block = 256 threads = 8 waves (4 M-tiles x 2 N-tiles of 32x32) -> 128x64 tile.
// reverse=1 gathers x rows time-reversed per batch (backward direction).
// ---------------------------------------------------------------------------
__global__ __launch_bounds__(256)
void gate_gemm_kernel(const bf16_t* __restrict__ xbf,
                      const bf16_t* __restrict__ Wzt,
                      const bf16_t* __restrict__ Wht,
                      const float* __restrict__ bz,
                      const float* __restrict__ bh,
                      float* __restrict__ lc,
                      float* __restrict__ lzh,
                      int reverse) {
    const int lane = threadIdx.x & 31;
    const int wv   = threadIdx.x >> 5;
    const int wm   = wv & 3, wn = wv >> 2;
    const int rowBase = blockIdx.y * 128 + wm * 32;
    const int colBase = blockIdx.x * 64 + wn * 32;
    const int kh = lane >> 4;

    // per-lane source rows (time-reversed per batch for backward pass)
    const bf16_t* arow[2];
    #pragma unroll
    for (int i = 0; i < 2; ++i) {
        int r = rowBase + i * 16 + (lane & 15);
        if (reverse) { int b = r >> 12; int s = r & (Ssz - 1); r = (b << 12) + (Ssz - 1 - s); }
        arow[i] = xbf + (size_t)r * Dsz;
    }

    v8f ak[2][2] = {};
    v8f ah[2][2] = {};

    for (int kt = 0; kt < Dsz; kt += 32) {
        v16bf a0 = load_fragA_row(arow[0], kt, kh);
        v16bf a1 = load_fragA_row(arow[1], kt, kh);
        v16bf z0 = load_fragB(Wzt, Dsz, colBase,      kt, lane);
        v16bf z1 = load_fragB(Wzt, Dsz, colBase + 16, kt, lane);
        v16bf h0 = load_fragB(Wht, Dsz, colBase,      kt, lane);
        v16bf h1 = load_fragB(Wht, Dsz, colBase + 16, kt, lane);
        ak[0][0] = wmma_bf16(a0, z0, ak[0][0]);
        ak[0][1] = wmma_bf16(a0, z1, ak[0][1]);
        ak[1][0] = wmma_bf16(a1, z0, ak[1][0]);
        ak[1][1] = wmma_bf16(a1, z1, ak[1][1]);
        ah[0][0] = wmma_bf16(a0, h0, ah[0][0]);
        ah[0][1] = wmma_bf16(a0, h1, ah[0][1]);
        ah[1][0] = wmma_bf16(a1, h0, ah[1][0]);
        ah[1][1] = wmma_bf16(a1, h1, ah[1][1]);
    }

    #pragma unroll
    for (int i = 0; i < 2; ++i) {
        #pragma unroll
        for (int j = 0; j < 2; ++j) {
            int n = colBase + j * 16 + (lane & 15);
            float bzv = bz[n], bhv = bh[n];
            int m0 = rowBase + i * 16 + 8 * kh;
            #pragma unroll
            for (int v = 0; v < 8; ++v) {
                int r = m0 + v;
                float kk = ak[i][j][v] + bzv;
                float hp = ah[i][j][v] + bhv;
                size_t idx = (size_t)r * Hsz + n;
                lc[idx]  = -softplusf(kk);
                lzh[idx] = -softplusf(-kk) + log_g(hp);
            }
        }
    }
}

// ---------------------------------------------------------------------------
// Chunked log-space scan: logh_t = logaddexp(lc_t + logh_{t-1}, lzh_t).
// Operator on (A=sum lc, B=logh):  (A1,B1) o (a,b) = (A1+a, logaddexp(a+B1, b))
// Pass1: per-chunk summaries.  Pass2: exclusive prefix over chunks.
// Pass3: re-scan applying carry, write h = exp(logh) as bf16 into hcat.
// ---------------------------------------------------------------------------
__global__ __launch_bounds__(256)
void scan_pass1_kernel(const float* __restrict__ lcF, const float* __restrict__ lzhF,
                       const float* __restrict__ lcB, const float* __restrict__ lzhB,
                       float2* __restrict__ summ) {
    int dir = blockIdx.z;
    const float* lc  = dir ? lcB  : lcF;
    const float* lzh = dir ? lzhB : lzhF;
    int c = blockIdx.x;
    int b = blockIdx.y >> 2;               // H/256 = 4 h-blocks
    int h = (blockIdx.y & 3) * 256 + threadIdx.x;
    size_t base = ((size_t)b * Ssz + (size_t)c * CHS) * Hsz + h;
    float A = 0.0f, LB = -INFINITY;
    for (int s = 0; s < CHS; ++s) {
        float a = lc[base + (size_t)s * Hsz];
        float z = lzh[base + (size_t)s * Hsz];
        LB = logaddexpf_(a + LB, z);
        A += a;
    }
    int ch = (dir * Bsz + b) * Hsz + h;
    summ[(size_t)ch * NCH + c] = make_float2(A, LB);
}

__global__ __launch_bounds__(256)
void scan_pass2_kernel(float2* __restrict__ summ) {
    int ch = blockIdx.x * blockDim.x + threadIdx.x;
    if (ch >= 2 * Bsz * Hsz) return;
    float cA = 0.0f, cB = -INFINITY;
    for (int c = 0; c < NCH; ++c) {
        float2 v = summ[(size_t)ch * NCH + c];
        summ[(size_t)ch * NCH + c] = make_float2(cA, cB);   // exclusive prefix
        float nB = logaddexpf_(v.x + cB, v.y);
        cA += v.x;
        cB = nB;
    }
}

__global__ __launch_bounds__(256)
void scan_pass3_kernel(const float* __restrict__ lcF, const float* __restrict__ lzhF,
                       const float* __restrict__ lcB, const float* __restrict__ lzhB,
                       const float2* __restrict__ summ, bf16_t* __restrict__ hcat) {
    int dir = blockIdx.z;
    const float* lc  = dir ? lcB  : lcF;
    const float* lzh = dir ? lzhB : lzhF;
    int c = blockIdx.x;
    int b = blockIdx.y >> 2;
    int h = (blockIdx.y & 3) * 256 + threadIdx.x;
    int ch = (dir * Bsz + b) * Hsz + h;
    float2 carry = summ[(size_t)ch * NCH + c];
    size_t base = ((size_t)b * Ssz + (size_t)c * CHS) * Hsz + h;
    float lA = 0.0f, lB = -INFINITY;
    for (int s = 0; s < CHS; ++s) {
        float a = lc[base + (size_t)s * Hsz];
        float z = lzh[base + (size_t)s * Hsz];
        lA += a;
        lB = logaddexpf_(a + lB, z);
        float lh = logaddexpf_(lA + carry.y, lB);
        size_t row = (size_t)b * Ssz + (size_t)c * CHS + s;
        hcat[row * K2 + (size_t)dir * Hsz + h] = (bf16_t)__expf(lh);
    }
}

// ---------------------------------------------------------------------------
// Output GEMM:  out = hcat(bf16 [M,2H]) @ W_out + b_out   (f32 out)
// ---------------------------------------------------------------------------
__global__ __launch_bounds__(256)
void out_gemm_kernel(const bf16_t* __restrict__ hcat,
                     const bf16_t* __restrict__ Woutt,
                     const float* __restrict__ bout,
                     float* __restrict__ out) {
    const int lane = threadIdx.x & 31;
    const int wv   = threadIdx.x >> 5;
    const int wm   = wv & 3, wn = wv >> 2;
    const int rowBase = blockIdx.y * 128 + wm * 32;
    const int colBase = blockIdx.x * 64 + wn * 32;
    const int kh = lane >> 4;

    const bf16_t* arow0 = hcat + (size_t)(rowBase + (lane & 15)) * K2;
    const bf16_t* arow1 = arow0 + (size_t)16 * K2;

    v8f acc[2][2] = {};
    for (int kt = 0; kt < K2; kt += 32) {
        v16bf a0 = load_fragA_row(arow0, kt, kh);
        v16bf a1 = load_fragA_row(arow1, kt, kh);
        v16bf b0 = load_fragB(Woutt, K2, colBase,      kt, lane);
        v16bf b1 = load_fragB(Woutt, K2, colBase + 16, kt, lane);
        acc[0][0] = wmma_bf16(a0, b0, acc[0][0]);
        acc[0][1] = wmma_bf16(a0, b1, acc[0][1]);
        acc[1][0] = wmma_bf16(a1, b0, acc[1][0]);
        acc[1][1] = wmma_bf16(a1, b1, acc[1][1]);
    }

    #pragma unroll
    for (int i = 0; i < 2; ++i) {
        #pragma unroll
        for (int j = 0; j < 2; ++j) {
            int n = colBase + j * 16 + (lane & 15);
            float bb = bout[n];
            int m0 = rowBase + i * 16 + 8 * kh;
            #pragma unroll
            for (int v = 0; v < 8; ++v)
                out[(size_t)(m0 + v) * Hsz + n] = acc[i][j][v] + bb;
        }
    }
}

// ---------------------------------------------------------------------------
// launch
// ---------------------------------------------------------------------------
extern "C" void kernel_launch(void* const* d_in, const int* in_sizes, int n_in,
                              void* d_out, int out_size, void* d_ws, size_t ws_size,
                              hipStream_t stream) {
    (void)in_sizes; (void)n_in; (void)out_size; (void)ws_size;

    const float* x     = (const float*)d_in[0];
    const float* Wz_f  = (const float*)d_in[1];
    const float* bz_f  = (const float*)d_in[2];
    const float* Wh_f  = (const float*)d_in[3];
    const float* bh_f  = (const float*)d_in[4];
    const float* Wz_b  = (const float*)d_in[5];
    const float* bz_b  = (const float*)d_in[6];
    const float* Wh_b  = (const float*)d_in[7];
    const float* bh_b  = (const float*)d_in[8];
    const float* W_out = (const float*)d_in[9];
    const float* b_out = (const float*)d_in[10];
    float* out = (float*)d_out;

    char* ws = (char*)d_ws;
    size_t off = 0;
    auto alloc = [&](size_t bytes) -> void* {
        off = (off + 255) & ~(size_t)255;
        void* p = ws + off;
        off += bytes;
        return p;
    };

    float*  lc_f  = (float*)alloc((size_t)Msz * Hsz * 4);
    float*  lzh_f = (float*)alloc((size_t)Msz * Hsz * 4);
    float*  lc_b  = (float*)alloc((size_t)Msz * Hsz * 4);
    float*  lzh_b = (float*)alloc((size_t)Msz * Hsz * 4);
    bf16_t* xbf   = (bf16_t*)alloc((size_t)Msz * Dsz * 2);
    bf16_t* Wzft  = (bf16_t*)alloc((size_t)Hsz * Dsz * 2);
    bf16_t* Whft  = (bf16_t*)alloc((size_t)Hsz * Dsz * 2);
    bf16_t* Wzbt  = (bf16_t*)alloc((size_t)Hsz * Dsz * 2);
    bf16_t* Whbt  = (bf16_t*)alloc((size_t)Hsz * Dsz * 2);
    bf16_t* Woutt = (bf16_t*)alloc((size_t)Hsz * K2  * 2);
    bf16_t* hcat  = (bf16_t*)alloc((size_t)Msz * K2  * 2);
    float2* summ  = (float2*)alloc((size_t)2 * Bsz * Hsz * NCH * 8);

    // 1. precision conversion / weight transpose
    {
        size_t n4 = (size_t)Msz * Dsz / 4;
        convert_bf16_kernel<<<dim3((unsigned)((n4 + 255) / 256)), dim3(256), 0, stream>>>(x, xbf, n4);
        dim3 tb(32, 8);
        transconv_kernel<<<dim3(Hsz / 32, Dsz / 32), tb, 0, stream>>>(Wz_f, Wzft, Dsz, Hsz);
        transconv_kernel<<<dim3(Hsz / 32, Dsz / 32), tb, 0, stream>>>(Wh_f, Whft, Dsz, Hsz);
        transconv_kernel<<<dim3(Hsz / 32, Dsz / 32), tb, 0, stream>>>(Wz_b, Wzbt, Dsz, Hsz);
        transconv_kernel<<<dim3(Hsz / 32, Dsz / 32), tb, 0, stream>>>(Wh_b, Whbt, Dsz, Hsz);
        transconv_kernel<<<dim3(Hsz / 32, K2 / 32), tb, 0, stream>>>(W_out, Woutt, K2, Hsz);
    }

    // 2. gate GEMMs (forward + reversed)
    {
        dim3 grid(Hsz / 64, Msz / 128);
        gate_gemm_kernel<<<grid, 256, 0, stream>>>(xbf, Wzft, Whft, bz_f, bh_f, lc_f, lzh_f, 0);
        gate_gemm_kernel<<<grid, 256, 0, stream>>>(xbf, Wzbt, Whbt, bz_b, bh_b, lc_b, lzh_b, 1);
    }

    // 3. chunked log-space scan -> hcat (bf16 [M, 2H])
    {
        dim3 grid1(NCH, Bsz * (Hsz / 256), 2);
        scan_pass1_kernel<<<grid1, 256, 0, stream>>>(lc_f, lzh_f, lc_b, lzh_b, summ);
        scan_pass2_kernel<<<dim3(2 * Bsz * Hsz / 256), 256, 0, stream>>>(summ);
        scan_pass3_kernel<<<grid1, 256, 0, stream>>>(lc_f, lzh_f, lc_b, lzh_b, summ, hcat);
    }

    // 4. output GEMM
    {
        dim3 grid(Hsz / 64, Msz / 128);
        out_gemm_kernel<<<grid, 256, 0, stream>>>(hcat, Woutt, b_out, out);
    }
}
